// EmbraceNet_74345883894015
// MI455X (gfx1250) — matrix operand order repaired and checked by reading, not compiled, not run on hardware
//
#include <hip/hip_runtime.h>

#define BATCH 4096
#define EMB   2048
#define INDIM 1024
#define NMOD  4

// Workgroup tile: 128 (M) x 128 (N), K-step 32 (bf16 WMMA K). 8 wave32 waves.
#define TM 128
#define TN 128
#define TK 32
#define KSTEPS (INDIM / TK)

typedef __attribute__((ext_vector_type(16))) __bf16 v16bf;
typedef __attribute__((ext_vector_type(8)))  __bf16 v8bf;
typedef __attribute__((ext_vector_type(2)))  __bf16 v2bf;
typedef __attribute__((ext_vector_type(8)))  float  v8f;
typedef __attribute__((ext_vector_type(2)))  float  vf2;

__device__ __forceinline__ unsigned int pack_bf2(float lo, float hi) {
    // float2 -> bf16x2 convertvector: lowers to a single v_cvt_pk_bf16_f32
    vf2 f; f.x = lo; f.y = hi;
    v2bf v = __builtin_convertvector(f, v2bf);
    return __builtin_bit_cast(unsigned int, v);
}

// Deterministic stand-in for the categorical modality pick: uniform 4-way hash.
__device__ __forceinline__ unsigned int pick4(unsigned int key) {
    unsigned int h = key * 2654435761u + 42u;
    h ^= h >> 16; h *= 0x85ebca6bu; h ^= h >> 13;
    return h & 3u;
}

// ---- Stage-in: buffer one K-step's A (128x32) and B (32x128) fp32 tiles in registers ----
__device__ __forceinline__ void load_tiles(const float* __restrict__ X,
                                           const float* __restrict__ W,
                                           int tileMbase, int tileNbase, int k0, int t,
                                           float4 ra[4], float4 rb[4]) {
    // A: thread t owns 16 contiguous floats of row (t>>1), half (t&1).
    const float* pa = X + (size_t)(tileMbase + (t >> 1)) * INDIM + k0 + (t & 1) * 16;
    #pragma unroll
    for (int i = 0; i < 4; ++i) ra[i] = *(const float4*)(pa + i * 4);
    __builtin_prefetch(pa + 2 * TK, 0, 1);       // speculative, 2 K-steps ahead

    // B: thread owns 4 consecutive K rows at the same 4 columns (coalesced 512B/row/wave)
    const int n4 = (t & 31) << 2;
    const int c0 = (t >> 5) << 2;
    #pragma unroll
    for (int i = 0; i < 4; ++i) {
        const float* p = W + (size_t)(k0 + c0 + i) * EMB + tileNbase + n4;
        rb[i] = *(const float4*)p;
        __builtin_prefetch(p + (size_t)(2 * TK) * EMB, 0, 1);
    }
}

// ---- Convert fp32 regs -> bf16 LDS (A row-major [m][k], B transposed [n][k]) ----
__device__ __forceinline__ void store_tiles(unsigned short* __restrict__ ldsA,
                                            unsigned short* __restrict__ ldsB,
                                            int t, const float4 ra[4], const float4 rb[4]) {
    // A: 16 bf16 -> two ds_store_b128
    unsigned short* da = ldsA + (t >> 1) * TK + (t & 1) * 16;
    uint4 p0, p1;
    p0.x = pack_bf2(ra[0].x, ra[0].y); p0.y = pack_bf2(ra[0].z, ra[0].w);
    p0.z = pack_bf2(ra[1].x, ra[1].y); p0.w = pack_bf2(ra[1].z, ra[1].w);
    p1.x = pack_bf2(ra[2].x, ra[2].y); p1.y = pack_bf2(ra[2].z, ra[2].w);
    p1.z = pack_bf2(ra[3].x, ra[3].y); p1.w = pack_bf2(ra[3].z, ra[3].w);
    *(uint4*)(da)     = p0;
    *(uint4*)(da + 8) = p1;

    // B transpose: column j gets 4 consecutive K values -> ds_store_b64
    const int n4 = (t & 31) << 2;
    const int c0 = (t >> 5) << 2;
    const float* fb = (const float*)rb;          // [i*4 + j] = (k=c0+i, n=n4+j)
    #pragma unroll
    for (int j = 0; j < 4; ++j) {
        uint2 pk;
        pk.x = pack_bf2(fb[0 * 4 + j], fb[1 * 4 + j]);
        pk.y = pack_bf2(fb[2 * 4 + j], fb[3 * 4 + j]);
        *(uint2*)(ldsB + (n4 + j) * TK + c0) = pk;
    }
}

// ---- One modality: K=1024 GEMM into registers, then predicated select-store ----
__device__ __forceinline__ void gemm_modality(const float* __restrict__ X,
                                              const float* __restrict__ W,
                                              const float* __restrict__ Bv,
                                              float* __restrict__ out, int f,
                                              unsigned short* __restrict__ ldsA,  // [2][TM*TK]
                                              unsigned short* __restrict__ ldsB)  // [2][TN*TK]
{
    const int t     = threadIdx.x;
    const int lane  = t & 31;
    const int wave  = t >> 5;
    const int waveM = wave & 3;      // 4 slabs of 32 rows
    const int waveN = wave >> 2;     // 2 slabs of 64 cols
    const int half  = lane >> 4;
    const int l16   = lane & 15;
    const int tileNbase = blockIdx.x * TN;
    const int tileMbase = blockIdx.y * TM;

    v8f acc[2][4];
    #pragma unroll
    for (int mi = 0; mi < 2; ++mi)
        #pragma unroll
        for (int ni = 0; ni < 4; ++ni)
            #pragma unroll
            for (int e = 0; e < 8; ++e) acc[mi][ni][e] = 0.0f;

    float4 ra[4], rb[4];
    load_tiles(X, W, tileMbase, tileNbase, 0, t, ra, rb);     // prologue

    int buf = 0;
    for (int kk = 0; kk < KSTEPS; ++kk) {
        unsigned short* lA = ldsA + buf * (TM * TK);
        unsigned short* lB = ldsB + buf * (TN * TK);
        store_tiles(lA, lB, t, ra, rb);
        __syncthreads();

        // pipeline: issue next K-step's global loads (clamped, branch-free last iter)
        const int knext = (kk + 1 < KSTEPS) ? (kk + 1) * TK : kk * TK;
        load_tiles(X, W, tileMbase, tileNbase, knext, t, ra, rb);

        // A fragments (lanes 0-15: K0-15, lanes 16-31: K16-31)
        v16bf afrag[2];
        #pragma unroll
        for (int mi = 0; mi < 2; ++mi) {
            const unsigned short* rp = lA + (waveM * 32 + mi * 16 + l16) * TK;
            afrag[mi] = *(const v16bf*)(rp + half * 16);
        }
        // B fragments (lanes 0-15: K0-15, lanes 16-31: K16-31; contiguous in [n][k] LDS)
        v16bf bfrag[4];
        #pragma unroll
        for (int ni = 0; ni < 4; ++ni)
            bfrag[ni] = *(const v16bf*)(lB + (waveN * 64 + ni * 16 + l16) * TK + half * 16);

        #pragma unroll
        for (int mi = 0; mi < 2; ++mi)
            #pragma unroll
            for (int ni = 0; ni < 4; ++ni)
                acc[mi][ni] = __builtin_amdgcn_wmma_f32_16x16x32_bf16(
                    false, afrag[mi], false, bfrag[ni],
                    (short)0, acc[mi][ni], false, false);

        buf ^= 1;
    }

    // Epilogue: bias + ReLU, store only where the hash picks modality f
    #pragma unroll
    for (int mi = 0; mi < 2; ++mi) {
        #pragma unroll
        for (int ni = 0; ni < 4; ++ni) {
            const int col  = tileNbase + waveN * 64 + ni * 16 + l16;
            const float bias = Bv[col];
            #pragma unroll
            for (int r = 0; r < 8; ++r) {
                const int row = tileMbase + waveM * 32 + mi * 16 + half * 8 + r;
                const unsigned int key = (unsigned int)row * (unsigned int)EMB + (unsigned int)col;
                if (pick4(key) == (unsigned int)f) {
                    float v = acc[mi][ni][r] + bias;
                    out[(size_t)row * EMB + col] = v > 0.0f ? v : 0.0f;
                }
            }
        }
    }
    // No extra barrier needed before the next modality: its first store to lds[0]
    // happens after this modality's sync at kk=31, which ordered all reads of lds[0].
}

__global__ __launch_bounds__(256)
void embrace_fused(const float* __restrict__ x0, const float* __restrict__ W0, const float* __restrict__ b0,
                   const float* __restrict__ x1, const float* __restrict__ W1, const float* __restrict__ b1,
                   const float* __restrict__ x2, const float* __restrict__ W2, const float* __restrict__ b2,
                   const float* __restrict__ x3, const float* __restrict__ W3, const float* __restrict__ b3,
                   float* __restrict__ out)
{
    // Double-buffered bf16 tiles: 2 x (8KB + 8KB) = 32KB of the WGP's 320KB LDS.
    __shared__ __align__(16) unsigned short ldsA[2 * TM * TK];
    __shared__ __align__(16) unsigned short ldsB[2 * TN * TK];

    // Direct kernel-arg pointers per call -> global address space inference (no FLAT).
    gemm_modality(x0, W0, b0, out, 0, ldsA, ldsB);
    gemm_modality(x1, W1, b1, out, 1, ldsA, ldsB);
    gemm_modality(x2, W2, b2, out, 2, ldsA, ldsB);
    gemm_modality(x3, W3, b3, out, 3, ldsA, ldsB);
}

extern "C" void kernel_launch(void* const* d_in, const int* in_sizes, int n_in,
                              void* d_out, int out_size, void* d_ws, size_t ws_size,
                              hipStream_t stream) {
    const float* x[NMOD]; const float* W[NMOD]; const float* b[NMOD];
    for (int i = 0; i < NMOD; ++i) {
        x[i] = (const float*)d_in[3 * i + 0];
        W[i] = (const float*)d_in[3 * i + 1];
        b[i] = (const float*)d_in[3 * i + 2];
    }
    dim3 grid(EMB / TN, BATCH / TM);   // 16 x 32 = 512 workgroups
    embrace_fused<<<grid, 256, 0, stream>>>(
        x[0], W[0], b[0], x[1], W[1], b[1],
        x[2], W[2], b[2], x[3], W[3], b[3],
        (float*)d_out);
}
